// ScaledDotProductAttention_33174327394411
// MI455X (gfx1250) — compile-verified, matrix operand order checked by reading, not run on hardware
//
#include <hip/hip_runtime.h>
#include <hip/hip_bf16.h>

typedef __attribute__((ext_vector_type(16))) __bf16 v16bf;
typedef __attribute__((ext_vector_type(8)))  __bf16 v8bf;
typedef __attribute__((ext_vector_type(4)))  __bf16 v4bf;
typedef __attribute__((ext_vector_type(8)))  float  v8f;

constexpr int B  = 2;
constexpr int H  = 16;
constexpr int S  = 2048;
constexpr int Dh = 64;
constexpr size_t NELEM = (size_t)B * H * S * Dh;   // 4,194,304 == S*S
constexpr float QSCALE = 0.125f * 1.44269504088896340736f;  // (1/temp)*log2(e)

#define NEGINF (-1.0e30f)

// ---------------------------------------------------------------------------
// Pass 1: packing into d_ws.
//   qb[bh][s][d] = q * (1/temp) * log2(e)  (bf16; softmax runs in exp2 domain)
//   kb[bh][s][d] = k                        (bf16 row-major)
//   vt[bh][d][s] = v                        (bf16 TRANSPOSED for PV B-frags)
//   mb[q][k]     = mask ? 0 : -1e30         (f32 additive bias; S*S == NELEM)
// ---------------------------------------------------------------------------
__global__ __launch_bounds__(256)
void pack_kernel(const float* __restrict__ q,
                 const float* __restrict__ k,
                 const float* __restrict__ v,
                 const int*   __restrict__ mask,
                 __bf16* __restrict__ qb,
                 __bf16* __restrict__ kb,
                 __bf16* __restrict__ vt,
                 float*  __restrict__ mb)
{
    const size_t i4 = ((size_t)blockIdx.x * blockDim.x + threadIdx.x) * 4;
    if (i4 >= NELEM) return;

    float4 xq = *reinterpret_cast<const float4*>(q + i4);
    float4 xk = *reinterpret_cast<const float4*>(k + i4);
    float4 xv = *reinterpret_cast<const float4*>(v + i4);
    int4   xm = *reinterpret_cast<const int4*>(mask + i4);

    v4bf yq = { (__bf16)(xq.x * QSCALE), (__bf16)(xq.y * QSCALE),
                (__bf16)(xq.z * QSCALE), (__bf16)(xq.w * QSCALE) };
    v4bf yk = { (__bf16)xk.x, (__bf16)xk.y, (__bf16)xk.z, (__bf16)xk.w };
    *reinterpret_cast<v4bf*>(qb + i4) = yq;
    *reinterpret_cast<v4bf*>(kb + i4) = yk;

    float4 yb = { xm.x ? 0.0f : NEGINF, xm.y ? 0.0f : NEGINF,
                  xm.z ? 0.0f : NEGINF, xm.w ? 0.0f : NEGINF };
    *reinterpret_cast<float4*>(mb + i4) = yb;

    const int    d  = (int)(i4 % Dh);
    const size_t s  = (i4 / Dh) % S;
    const size_t bh = i4 / ((size_t)Dh * S);
    __bf16* __restrict__ vcol = vt + (bh * Dh) * S + s;
    vcol[(size_t)(d + 0) * S] = (__bf16)xv.x;
    vcol[(size_t)(d + 1) * S] = (__bf16)xv.y;
    vcol[(size_t)(d + 2) * S] = (__bf16)xv.z;
    vcol[(size_t)(d + 3) * S] = (__bf16)xv.w;
}

// 16x32 bf16 fragment, CDNA5 16-bit striped layout: per lane two contiguous
// 8-element groups at +0..7 and +16..23 (hi*8 half-wave offset baked into p).
__device__ __forceinline__ v16bf load_frag(const __bf16* __restrict__ p) {
    v8bf a = *reinterpret_cast<const v8bf*>(p);
    v8bf b = *reinterpret_cast<const v8bf*>(p + 16);
    v16bf f;
#pragma unroll
    for (int e = 0; e < 8; ++e) { f[e] = a[e]; f[e + 8] = b[e]; }
    return f;
}

// ---------------------------------------------------------------------------
// Pass 2: flash attention. One wave32 owns 16 query rows; 8 waves / block.
// Scores computed TRANSPOSED (A = K-tile, B = Q^T) so the C-layout result is
// already the A-layout for P x V.  Mask enters as the WMMA C operand; softmax
// runs in exp2 domain; stats update once per 64 keys.
// ---------------------------------------------------------------------------
__global__ __launch_bounds__(256, 2)
void fattn_bf16_wmma_kernel(const __bf16* __restrict__ qb,
                            const __bf16* __restrict__ kb,
                            const __bf16* __restrict__ vt,
                            const float*  __restrict__ mb,
                            float*        __restrict__ out)
{
    const int lane = threadIdx.x & 31;
    const int wave = threadIdx.x >> 5;
    const int hi   = lane >> 4;
    const int ln   = lane & 15;

    constexpr int QTILES_PER_HEAD = S / 128;
    const int bh    = blockIdx.x / QTILES_PER_HEAD;
    const int qblk  = blockIdx.x % QTILES_PER_HEAD;
    const int qbase = qblk * 128 + wave * 16;

    const __bf16* __restrict__ Qp = qb + ((size_t)bh * S + qbase) * Dh;
    const __bf16* __restrict__ Kp = kb + (size_t)bh * S * Dh;
    const __bf16* __restrict__ Vt = vt + (size_t)bh * Dh * S;
    float*        __restrict__ Op = out + ((size_t)bh * S + qbase) * Dh;
    const float*  __restrict__ Brow = mb + (size_t)(qbase + ln) * S + hi * 8;

    v16bf qf[2];
#pragma unroll
    for (int c = 0; c < 2; ++c)
        qf[c] = load_frag(Qp + ln * Dh + c * 32 + hi * 8);

    const __bf16* __restrict__ Krow = Kp + (size_t)ln * Dh + hi * 8;
    const __bf16* __restrict__ Vrow[4] = {
        Vt + (size_t)(0 * 16 + ln) * S + hi * 8,
        Vt + (size_t)(1 * 16 + ln) * S + hi * 8,
        Vt + (size_t)(2 * 16 + ln) * S + hi * 8,
        Vt + (size_t)(3 * 16 + ln) * S + hi * 8,
    };

    v8f acc[4];
#pragma unroll
    for (int c = 0; c < 4; ++c)
#pragma unroll
        for (int r = 0; r < 8; ++r) acc[c][r] = 0.0f;

    float m_run = NEGINF;
    float l_run = 0.0f;

    for (int k0 = 0; k0 < S; k0 += 64) {
        if (k0 + 64 < S) {
            __builtin_prefetch(Krow + (size_t)(k0 + 64) * Dh, 0, 3);
            __builtin_prefetch(Vrow[0] + (k0 + 64), 0, 3);
        }

        // ---- scores^T: four 16-key tiles, mask bias as the C operand ----
        v8f st[4];
#pragma unroll
        for (int t = 0; t < 4; ++t) {
            const int kb16 = k0 + t * 16;
            float4 b0 = *reinterpret_cast<const float4*>(Brow + kb16);
            float4 b1 = *reinterpret_cast<const float4*>(Brow + kb16 + 4);
            v8f c8 = {b0.x, b0.y, b0.z, b0.w, b1.x, b1.y, b1.z, b1.w};
#pragma unroll
            for (int c = 0; c < 2; ++c) {
                v16bf ka = load_frag(Krow + (size_t)kb16 * Dh + c * 32);
                c8 = __builtin_amdgcn_wmma_f32_16x16x32_bf16(
                        false, ka, false, qf[c], (short)0, c8, false, false);
            }
            st[t] = c8;
        }

        // ---- online softmax over 64 keys (exp2 domain, per-lane query) ----
        float rmax = NEGINF;
#pragma unroll
        for (int t = 0; t < 4; ++t)
#pragma unroll
            for (int r = 0; r < 8; ++r) rmax = fmaxf(rmax, st[t][r]);
        rmax = fmaxf(rmax, __shfl_xor(rmax, 16, 32));

        const float mnew  = fmaxf(m_run, rmax);
        const float alpha = __builtin_amdgcn_exp2f(m_run - mnew);

        v16bf pa[2];   // A-layout P fragments: pa[s] covers keys k0+s*32..+31
        float rsum = 0.0f;
#pragma unroll
        for (int s = 0; s < 2; ++s)
#pragma unroll
            for (int t = 0; t < 2; ++t)
#pragma unroll
                for (int r = 0; r < 8; ++r) {
                    float p = __builtin_amdgcn_exp2f(st[s * 2 + t][r] - mnew);
                    rsum += p;
                    pa[s][t * 8 + r] = (__bf16)p;
                }
        rsum += __shfl_xor(rsum, 16, 32);

        l_run = l_run * alpha + rsum;
        m_run = mnew;

        // ---- rescale accumulators once per 64 keys ----
        float ar[8];
#pragma unroll
        for (int r = 0; r < 8; ++r) ar[r] = __shfl(alpha, r + hi * 8, 32);
#pragma unroll
        for (int c = 0; c < 4; ++c)
#pragma unroll
            for (int r = 0; r < 8; ++r) acc[c][r] *= ar[r];

        // ---- P x V: two 32-key steps x four 16-dim chunks ----
#pragma unroll
        for (int s = 0; s < 2; ++s)
#pragma unroll
            for (int c = 0; c < 4; ++c) {
                v16bf vb = load_frag(Vrow[c] + k0 + s * 32);
                acc[c] = __builtin_amdgcn_wmma_f32_16x16x32_bf16(
                            false, pa[s], false, vb, (short)0, acc[c], false, false);
            }
    }

    // ---- epilogue ----
    const float linv = 1.0f / l_run;
    float lr[8];
#pragma unroll
    for (int r = 0; r < 8; ++r) lr[r] = __shfl(linv, r + hi * 8, 32);
#pragma unroll
    for (int c = 0; c < 4; ++c)
#pragma unroll
        for (int r = 0; r < 8; ++r)
            Op[(size_t)(r + hi * 8) * Dh + c * 16 + ln] = acc[c][r] * lr[r];
}

extern "C" void kernel_launch(void* const* d_in, const int* in_sizes, int n_in,
                              void* d_out, int out_size, void* d_ws, size_t ws_size,
                              hipStream_t stream) {
    (void)in_sizes; (void)n_in; (void)out_size; (void)ws_size;
    const float* q    = (const float*)d_in[0];
    const float* k    = (const float*)d_in[1];
    const float* v    = (const float*)d_in[2];
    const int*   mask = (const int*)d_in[3];
    float*       out  = (float*)d_out;

    __bf16* qb = (__bf16*)d_ws;
    __bf16* kb = qb + NELEM;
    __bf16* vt = kb + NELEM;
    float*  mbias = (float*)(vt + NELEM);

    {
        const size_t threads = NELEM / 4;
        const int blk = 256;
        pack_kernel<<<(int)((threads + blk - 1) / blk), blk, 0, stream>>>(
            q, k, v, mask, qb, kb, vt, mbias);
    }
    {
        const int blocks = B * H * (S / 128);  // 512
        fattn_bf16_wmma_kernel<<<blocks, 256, 0, stream>>>(qb, kb, vt, mbias, out);
    }
}
